// Decoder_90855738179843
// MI455X (gfx1250) — matrix operand order repaired and checked by reading, not compiled, not run on hardware
//
#include <hip/hip_runtime.h>
#include <hip/hip_bf16.h>

// GRU decoder rollout on MI455X (gfx1250), persistent kernel, bf16 WMMA.
// Algebraic collapse: x(t+1)=y(t)=h(t)@W_head^T+b_head  =>
//   gi(t+1) = h(t) @ (W_ih*W_head)^T + (b_ih + W_ih*b_head)
// so r,z gates fuse with W_hh into one K=1024 GEMM and the head GEMM of step
// t-1 is independent of phase 1 of step t -> 2 grid barriers per step.
#define I_SZ 256
#define H_SZ 1024
#define B_SZ 512
#define L_SZ 256

#define KT_H   (H_SZ / 32)   // 32 k-tiles (K=1024 for every GEMM now)
#define NP1    1024          // 16 row-pairs x 64 col-strips (each 32x64 macro-tile)
#define NHEAD  64            // 16 row-pairs x 4 col-strips of the head GEMM
#define NUM_WG 128
#define TPB    256           // 8 wave32s per WG -> 1024 waves total

typedef __attribute__((ext_vector_type(16))) __bf16 v16bf;
typedef __attribute__((ext_vector_type(8)))  float  v8f;

static __device__ __forceinline__ v8f wmma_bf16(v16bf a, v16bf b, v8f c) {
  return __builtin_amdgcn_wmma_f32_16x16x32_bf16(
      false, a, false, b, (short)0, c, false, false);
}

// Pre-packed fragments: block = one 16x32 (A) / 32x16 (B) tile as [32 lanes][16 bf16],
// so each lane's fragment is one contiguous 32-byte load.
static __device__ __forceinline__ v16bf ldfrag(const __bf16* __restrict__ base,
                                               int blk, int lane) {
  return *(const v16bf*)(base + (((blk << 5) + lane) << 4));
}

// Inverse of ISA 16-bit A-matrix 16x32 layout: lane = M + 16*((K%16)/8),
// elem = (K%8) + 8*(K/16).
static __device__ __forceinline__ int a_pack_idx(int bt, int kt, int M, int K) {
  int half  = (K & 15) >> 3;
  int lanep = M + 16 * half;
  int e     = (K & 7) + 8 * (K >> 4);
  return ((((bt * KT_H + kt) << 5) + lanep) << 4) + e;
}

// Device-wide barrier with monotone counter (no reset races; zeroed per launch).
static __device__ __forceinline__ void grid_barrier(unsigned* bar, unsigned target) {
  __threadfence();
  __syncthreads();
  if (threadIdx.x == 0) {
    atomicAdd(bar, 1u);
    while (atomicAdd(bar, 0u) < target) __builtin_amdgcn_s_sleep(1);
  }
  __syncthreads();
  __threadfence();
}

// 32x64 macro-tile GEMM: 2 A-tiles x 4 B-tiles, K=1024 (32 k-steps).
// 6 fragment loads per 8 WMMAs. Adds bias[bias0 + col] and stores fp32.
static __device__ __forceinline__ void gemm_strip_2x4(
    const __bf16* __restrict__ Ap, int bt0,
    const __bf16* __restrict__ Bp, int ot0,
    const float* __restrict__ bias, int bias0,
    float* __restrict__ dst, int ldd, int col0, int lane) {
  v8f a00 = {}, a01 = {}, a02 = {}, a03 = {};
  v8f a10 = {}, a11 = {}, a12 = {}, a13 = {};
#pragma unroll 4
  for (int kt = 0; kt < KT_H; ++kt) {
    v16bf fa0 = ldfrag(Ap, (bt0    ) * KT_H + kt, lane);
    v16bf fa1 = ldfrag(Ap, (bt0 + 1) * KT_H + kt, lane);
    v16bf fb0 = ldfrag(Bp, (ot0    ) * KT_H + kt, lane);
    v16bf fb1 = ldfrag(Bp, (ot0 + 1) * KT_H + kt, lane);
    v16bf fb2 = ldfrag(Bp, (ot0 + 2) * KT_H + kt, lane);
    v16bf fb3 = ldfrag(Bp, (ot0 + 3) * KT_H + kt, lane);
    a00 = wmma_bf16(fa0, fb0, a00);  a01 = wmma_bf16(fa0, fb1, a01);
    a02 = wmma_bf16(fa0, fb2, a02);  a03 = wmma_bf16(fa0, fb3, a03);
    a10 = wmma_bf16(fa1, fb0, a10);  a11 = wmma_bf16(fa1, fb1, a11);
    a12 = wmma_bf16(fa1, fb2, a12);  a13 = wmma_bf16(fa1, fb3, a13);
  }
  int N = lane & 15, hh = lane >> 4;
  v8f* accs[8] = {&a00, &a01, &a02, &a03, &a10, &a11, &a12, &a13};
#pragma unroll
  for (int q = 0; q < 8; ++q) {
    int ri = q >> 2, cj = q & 3;
    float bv = bias[bias0 + cj * 16 + N];
    float* d = dst + (size_t)((bt0 + ri) * 16 + 8 * hh) * ldd + col0 + cj * 16 + N;
    v8f acc = *accs[q];
#pragma unroll
    for (int v = 0; v < 8; ++v) d[(size_t)v * ldd] = acc[v] + bv;
  }
}

// ---- one-time prep kernels --------------------------------------------------

// Wc = W_ih @ W_head  ([3H x I] @ [I x H] -> [3H x H]), fp32.
__global__ void comb_matmul(const float* __restrict__ W_ih,
                            const float* __restrict__ W_head,
                            float* __restrict__ Wc) {
  int stride = gridDim.x * blockDim.x;
  for (int idx = blockIdx.x * blockDim.x + threadIdx.x; idx < 3 * H_SZ * H_SZ;
       idx += stride) {
    int o = idx >> 10, j = idx & (H_SZ - 1);
    const float* wi = W_ih + (size_t)o * I_SZ;
    float s = 0.0f;
#pragma unroll 4
    for (int k = 0; k < I_SZ; ++k) s = fmaf(wi[k], W_head[(size_t)k * H_SZ + j], s);
    Wc[idx] = s;
  }
}

// brz0 = b_ih+b_hh (step 0); brz1 = brz0 + (W_ih@b_head) for rows<2H;
// bin1 = b_ih_n + (W_ih@b_head)_n.
__global__ void make_bias(const float* __restrict__ b_ih, const float* __restrict__ b_hh,
                          const float* __restrict__ W_ih, const float* __restrict__ b_head,
                          float* __restrict__ brz0, float* __restrict__ brz1,
                          float* __restrict__ bin1) {
  int o = blockIdx.x * blockDim.x + threadIdx.x;
  if (o >= 3 * H_SZ) return;
  float wib = 0.0f;
#pragma unroll 4
  for (int k = 0; k < I_SZ; ++k) wib = fmaf(W_ih[(size_t)o * I_SZ + k], b_head[k], wib);
  if (o < 2 * H_SZ) {
    float b0 = b_ih[o] + b_hh[o];
    brz0[o] = b0;
    brz1[o] = b0 + wib;
  } else {
    bin1[o - 2 * H_SZ] = b_ih[o] + wib;
  }
}

// Pack rows [rowOff, rowOff+16*OT) of (W [+ W2]) (both [? x 1024] fp32) into
// bf16 B-fragment blocks. B layout: lane l holds col N=l%16, K = (e%8) + 8*(l/16) + 16*(e/8).
__global__ void pack_weightB(const float* __restrict__ W, const float* __restrict__ W2,
                             __bf16* __restrict__ out, int OT, int rowOff) {
  int total = OT * KT_H * 512;
  for (int idx = blockIdx.x * blockDim.x + threadIdx.x; idx < total;
       idx += gridDim.x * blockDim.x) {
    int e     = idx & 15;
    int lanep = (idx >> 4) & 31;
    int blk   = idx >> 9;
    int kt    = blk & 31;
    int ot    = blk >> 5;
    int N     = lanep & 15;
    int half  = lanep >> 4;
    int k     = kt * 32 + (e & 7) + 8 * half + 16 * (e >> 3);
    int o     = rowOff + ot * 16 + N;
    float v = W[(size_t)o * H_SZ + k];
    if (W2) v += W2[(size_t)o * H_SZ + k];
    out[idx] = (__bf16)v;
  }
}

__global__ void init_state(const float* __restrict__ ctx, float* __restrict__ h_f32,
                           __bf16* __restrict__ h_p, unsigned* __restrict__ bar) {
  int gtid = blockIdx.x * blockDim.x + threadIdx.x;
  int stride = gridDim.x * blockDim.x;
  if (gtid == 0) *bar = 0u;
  for (int i = gtid; i < B_SZ * H_SZ; i += stride) {
    int b = i >> 10, j = i & (H_SZ - 1);
    float v = ctx[i];
    h_f32[i] = v;
    h_p[a_pack_idx(b >> 4, j >> 5, b & 15, j & 31)] = (__bf16)v;
  }
}

// ---- persistent rollout kernel ---------------------------------------------

__global__ void __launch_bounds__(TPB, 1)
gru_persistent(const __bf16* __restrict__ Wrz_p,  // (Wc+W_hh) rows 0..2H,   128 ot
               const __bf16* __restrict__ Wcn_p,  // Wc rows 2H..3H,          64 ot
               const __bf16* __restrict__ Whh_p,  // W_hh all rows,          192 ot
               const __bf16* __restrict__ Whd_p,  // W_head,                  16 ot
               const float* __restrict__ brz0, const float* __restrict__ brz1,
               const float* __restrict__ bin1,
               const float* __restrict__ b_ih, const float* __restrict__ b_hh,
               const float* __restrict__ b_head,
               float* __restrict__ h_f32, __bf16* __restrict__ h_p,
               float* __restrict__ Grz, float* __restrict__ Gin,
               float* __restrict__ Ghn,
               float* __restrict__ out, unsigned* __restrict__ bar) {
  const int lane     = threadIdx.x & 31;
  const int gwave    = blockIdx.x * (TPB / 32) + (threadIdx.x >> 5);
  const int nwaves   = gridDim.x * (TPB / 32);
  const int gtid     = blockIdx.x * TPB + threadIdx.x;
  const int nthreads = gridDim.x * TPB;
  unsigned  round    = 0;

  for (int t = 0; t < L_SZ; ++t) {
    // ---- Phase 1: gate GEMMs for step t, fused with head GEMM of step t-1.
    const int ntasks = (t == 0) ? NP1 : (NP1 + NHEAD);
    for (int task = gwave; task < ntasks; task += nwaves) {
      if (task < NP1) {
        int p = task >> 6, s = task & 63;
        int bt0 = p << 1;
        if (s < 32) {                       // fused r,z columns [0,2H)
          int ot0 = s << 2;
          if (t == 0)
            gemm_strip_2x4(h_p, bt0, Whh_p, ot0, brz0, ot0 * 16,
                           Grz, 2 * H_SZ, ot0 * 16, lane);
          else
            gemm_strip_2x4(h_p, bt0, Wrz_p, ot0, brz1, ot0 * 16,
                           Grz, 2 * H_SZ, ot0 * 16, lane);
        } else if (s < 48) {                // gi_n columns
          int sl = s - 32, ot0 = sl << 2;
          if (t == 0) {                     // x0==0 -> gi_n is just b_ih_n
            int N = lane & 15, hh = lane >> 4;
#pragma unroll
            for (int cj = 0; cj < 4; ++cj) {
              float bv = b_ih[2 * H_SZ + ot0 * 16 + cj * 16 + N];
#pragma unroll
              for (int ri = 0; ri < 2; ++ri) {
                float* d = Gin + (size_t)((bt0 + ri) * 16 + 8 * hh) * H_SZ
                               + ot0 * 16 + cj * 16 + N;
#pragma unroll
                for (int v = 0; v < 8; ++v) d[(size_t)v * H_SZ] = bv;
              }
            }
          } else {
            gemm_strip_2x4(h_p, bt0, Wcn_p, ot0, bin1, ot0 * 16,
                           Gin, H_SZ, ot0 * 16, lane);
          }
        } else {                            // gh_n columns (W_hh rows 2H..3H)
          int sl = s - 48, ot0 = sl << 2;
          gemm_strip_2x4(h_p, bt0, Whh_p, 128 + ot0, b_hh, 2 * H_SZ + ot0 * 16,
                         Ghn, H_SZ, ot0 * 16, lane);
        }
      } else {                              // head GEMM for step t-1
        int ht = task - NP1;
        int p = ht >> 2, sj = ht & 3;
        gemm_strip_2x4(h_p, p << 1, Whd_p, sj << 2, b_head, sj * 64,
                       out + (size_t)(t - 1) * B_SZ * I_SZ, I_SZ, sj * 64, lane);
      }
    }
    grid_barrier(bar, ++round * gridDim.x);

    // ---- Phase 2: gates elementwise; write fp32 h and A-packed bf16 h. ----
    for (int i = gtid; i < B_SZ * H_SZ; i += nthreads) {
      int b = i >> 10, j = i & (H_SZ - 1);
      float gr = Grz[(size_t)b * (2 * H_SZ) + j];
      float gz = Grz[(size_t)b * (2 * H_SZ) + H_SZ + j];
      float r  = 1.0f / (1.0f + __expf(-gr));
      float z  = 1.0f / (1.0f + __expf(-gz));
      float nn = tanhf(Gin[(size_t)b * H_SZ + j] + r * Ghn[(size_t)b * H_SZ + j]);
      float hn = (1.0f - z) * nn + z * h_f32[i];
      h_f32[i] = hn;
      h_p[a_pack_idx(b >> 4, j >> 5, b & 15, j & 31)] = (__bf16)hn;
    }
    grid_barrier(bar, ++round * gridDim.x);
  }

  // ---- Final head GEMM for step L-1 (stores complete before endpgm). ----
  for (int task = gwave; task < NHEAD; task += nwaves) {
    int p = task >> 2, sj = task & 3;
    gemm_strip_2x4(h_p, p << 1, Whd_p, sj << 2, b_head, sj * 64,
                   out + (size_t)(L_SZ - 1) * B_SZ * I_SZ, I_SZ, sj * 64, lane);
  }
}

extern "C" void kernel_launch(void* const* d_in, const int* in_sizes, int n_in,
                              void* d_out, int out_size, void* d_ws, size_t ws_size,
                              hipStream_t stream) {
  const float* ctx    = (const float*)d_in[0];
  // d_in[1] = length (constant 256, baked in)
  const float* W_ih   = (const float*)d_in[2];   // [3H, I]
  const float* W_hh   = (const float*)d_in[3];   // [3H, H]
  const float* b_ih   = (const float*)d_in[4];   // [3H]
  const float* b_hh   = (const float*)d_in[5];   // [3H]
  const float* W_head = (const float*)d_in[6];   // [I, H]
  const float* b_head = (const float*)d_in[7];   // [I]
  float* out = (float*)d_out;                    // [L, B, I]

  char* ws = (char*)d_ws;
  size_t off = 0;
  auto take = [&](size_t bytes) -> void* {
    void* p = ws + off;
    off = (off + bytes + 255) & ~(size_t)255;
    return p;
  };
  float*   Wc    = (float*) take((size_t)3 * H_SZ * H_SZ * 4);      // 12 MB
  __bf16*  Wrz_p = (__bf16*)take((size_t)128 * KT_H * 512 * 2);     // 4 MB
  __bf16*  Wcn_p = (__bf16*)take((size_t)64  * KT_H * 512 * 2);     // 2 MB
  __bf16*  Whh_p = (__bf16*)take((size_t)192 * KT_H * 512 * 2);     // 6 MB
  __bf16*  Whd_p = (__bf16*)take((size_t)16  * KT_H * 512 * 2);     // 0.5 MB
  float*   brz0  = (float*) take((size_t)2 * H_SZ * 4);
  float*   brz1  = (float*) take((size_t)2 * H_SZ * 4);
  float*   bin1  = (float*) take((size_t)H_SZ * 4);
  float*   h_f32 = (float*) take((size_t)B_SZ * H_SZ * 4);
  __bf16*  h_p   = (__bf16*)take((size_t)B_SZ * H_SZ * 2);
  float*   Grz   = (float*) take((size_t)B_SZ * 2 * H_SZ * 4);
  float*   Gin   = (float*) take((size_t)B_SZ * H_SZ * 4);
  float*   Ghn   = (float*) take((size_t)B_SZ * H_SZ * 4);
  unsigned* bar  = (unsigned*)take(256);

  hipLaunchKernelGGL(comb_matmul, dim3(2048), dim3(256), 0, stream, W_ih, W_head, Wc);
  hipLaunchKernelGGL(make_bias, dim3(12), dim3(256), 0, stream,
                     b_ih, b_hh, W_ih, b_head, brz0, brz1, bin1);
  hipLaunchKernelGGL(pack_weightB, dim3(768), dim3(256), 0, stream,
                     Wc, W_hh, Wrz_p, 128, 0);
  hipLaunchKernelGGL(pack_weightB, dim3(768), dim3(256), 0, stream,
                     Wc, (const float*)nullptr, Wcn_p, 64, 2 * H_SZ);
  hipLaunchKernelGGL(pack_weightB, dim3(768), dim3(256), 0, stream,
                     W_hh, (const float*)nullptr, Whh_p, 192, 0);
  hipLaunchKernelGGL(pack_weightB, dim3(768), dim3(256), 0, stream,
                     W_head, (const float*)nullptr, Whd_p, 16, 0);
  hipLaunchKernelGGL(init_state, dim3(512), dim3(256), 0, stream, ctx, h_f32, h_p, bar);
  hipLaunchKernelGGL(gru_persistent, dim3(NUM_WG), dim3(TPB), 0, stream,
                     Wrz_p, Wcn_p, Whh_p, Whd_p, brz0, brz1, bin1,
                     b_ih, b_hh, b_head, h_f32, h_p, Grz, Gin, Ghn, out, bar);
}
